// GATModel_8564164788972
// MI455X (gfx1250) — compile-verified, matrix operand order checked by reading, not run on hardware
//
#include <hip/hip_runtime.h>
#include <math.h>

typedef __attribute__((ext_vector_type(16))) _Float16 v16h;
typedef __attribute__((ext_vector_type(8)))  float    v8f;

#define NEG_SLOPE 0.2f
#define LN_EPS    1e-5f

__device__ __forceinline__ int f32_to_ordkey(float f) {
  int b = __float_as_int(f);
  return b >= 0 ? b : (b ^ 0x7FFFFFFF);
}
__device__ __forceinline__ float ordkey_to_f32(int k) {
  return __int_as_float(k < 0 ? (k ^ 0x7FFFFFFF) : k);
}

// ---------------------------------------------------------------------------
// x = concat(user_table, item_table)   [N,128] f32
// ---------------------------------------------------------------------------
__global__ void concat_kernel(const float* __restrict__ ut,
                              const float* __restrict__ it,
                              float* __restrict__ X, int num_users, int N) {
  int t = blockIdx.x * blockDim.x + threadIdx.x;
  int total = N * 32;                    // float4 granules
  if (t >= total) return;
  int node = t >> 5;
  int c4   = t & 31;
  const float4* srcp = (node < num_users)
      ? (const float4*)(ut + (size_t)node * 128)
      : (const float4*)(it + (size_t)(node - num_users) * 128);
  ((float4*)(X + (size_t)node * 128))[c4] = srcp[c4];
}

// ---------------------------------------------------------------------------
// zero acc [N*128], zero ssum [N*heads], mmax = INT_MIN [N*heads]
// ---------------------------------------------------------------------------
__global__ void init_kernel(float* __restrict__ acc, float* __restrict__ ssum,
                            int* __restrict__ mmax, int N, int heads) {
  int t = blockIdx.x * blockDim.x + threadIdx.x;
  if (t < N * 128) acc[t] = 0.0f;
  if (t < N * heads) { ssum[t] = 0.0f; mmax[t] = (int)0x80000000; }
}

// ---------------------------------------------------------------------------
// H = X @ W   (X:[N,128] f32, W:[128,128] f32, H:[N,128] f32) via WMMA f16
// block = 256 threads (8 waves); block does 16 rows x 128 cols; wave w -> 16 cols
// W staged TRANSPOSED in LDS (WT[col][k]) so both A and B fragments are
// contiguous 8-halfword runs at 16B-aligned offsets -> ds_load_b128.
// Bounds handling hoisted to uniform (per-block) branches.
// ---------------------------------------------------------------------------
__global__ __launch_bounds__(256) void gemm128_kernel(
    const float* __restrict__ X, const float* __restrict__ W,
    float* __restrict__ H, int N) {
  __shared__ _Float16 Alds[16][136];    // 272B row stride (16B multiple)
  __shared__ _Float16 WT[128][136];     // WT[col][k], 272B row stride

  int tid = threadIdx.x;
  int rowBase = blockIdx.x * 16;
  bool fullTile = (rowBase + 16 <= N);

#pragma unroll
  for (int i = 0; i < 64; ++i) {        // stage W transposed: 128x128
    int idx = tid + i * 256;
    int k = idx >> 7, col = idx & 127;  // global read coalesced
    WT[col][k] = (_Float16)W[idx];
  }
  if (fullTile) {
#pragma unroll
    for (int i = 0; i < 8; ++i) {       // 16x128 A tile, no clamping needed
      int idx = tid + i * 256;
      int r = idx >> 7, col = idx & 127;
      Alds[r][col] = (_Float16)X[(size_t)(rowBase + r) * 128 + col];
    }
  } else {
#pragma unroll
    for (int i = 0; i < 8; ++i) {
      int idx = tid + i * 256;
      int r = idx >> 7, col = idx & 127;
      int gr = rowBase + r; if (gr >= N) gr = N - 1;
      Alds[r][col] = (_Float16)X[(size_t)gr * 128 + col];
    }
  }
  __syncthreads();

  int wave = tid >> 5, lane = tid & 31;
  int m = lane & 15, kb = (lane >> 4) * 8, n = lane & 15;
  int colBase = wave * 16;

  v8f acc = {};
#pragma unroll
  for (int ks = 0; ks < 4; ++ks) {
    int k0 = ks * 32;
    v16h a, b;
#pragma unroll
    for (int i = 0; i < 8; ++i) {
      a[i]     = Alds[m][k0 + kb + i];
      a[i + 8] = Alds[m][k0 + kb + 16 + i];
      b[i]     = WT[colBase + n][k0 + kb + i];
      b[i + 8] = WT[colBase + n][k0 + kb + 16 + i];
    }
    acc = __builtin_amdgcn_wmma_f32_16x16x32_f16(false, a, false, b,
                                                 (short)0, acc, false, false);
  }

  // C layout: VGPR i -> row M = i + 8*(lane>=16), col N = lane&15
  float* hp = H + (size_t)(rowBase + kb) * 128 + colBase + n;
  if (fullTile) {
#pragma unroll
    for (int i = 0; i < 8; ++i) hp[(size_t)i * 128] = acc[i];
  } else {
#pragma unroll
    for (int i = 0; i < 8; ++i) {
      if (rowBase + kb + i < N) hp[(size_t)i * 128] = acc[i];
    }
  }
}

// ---------------------------------------------------------------------------
// a_s[n,h] = sum_c H[n,h,c]*att_s[h,c];  a_d likewise.  One wave per node.
// ---------------------------------------------------------------------------
__global__ __launch_bounds__(256) void att_coef_kernel(
    const float* __restrict__ H, const float* __restrict__ att_s,
    const float* __restrict__ att_d, float* __restrict__ a_s,
    float* __restrict__ a_d, int N, int heads) {
  int wid  = (blockIdx.x * blockDim.x + threadIdx.x) >> 5;
  int lane = threadIdx.x & 31;
  if (wid >= N) return;
  float4 hv = ((const float4*)(H + (size_t)wid * 128))[lane];
  float4 s4 = ((const float4*)att_s)[lane];
  float4 d4 = ((const float4*)att_d)[lane];
  float ps = hv.x * s4.x + hv.y * s4.y + hv.z * s4.z + hv.w * s4.w;
  float pd = hv.x * d4.x + hv.y * d4.y + hv.z * d4.z + hv.w * d4.w;
  int lph = 32 / heads;                 // lanes per head: 8 (H=4) or 32 (H=1)
  for (int msk = 1; msk < lph; msk <<= 1) {
    ps += __shfl_xor(ps, msk, 32);
    pd += __shfl_xor(pd, msk, 32);
  }
  int head = lane / lph;
  if ((lane % lph) == 0) {
    a_s[wid * heads + head] = ps;
    a_d[wid * heads + head] = pd;
  }
}

// ---------------------------------------------------------------------------
// per (edge,head): logit = leaky_relu(a_s[src]+a_d[dst]); atomicMax key per dst
// ---------------------------------------------------------------------------
__global__ void edge_max_kernel(const int* __restrict__ srcE,
                                const int* __restrict__ dstE,
                                const float* __restrict__ a_s,
                                const float* __restrict__ a_d,
                                float* __restrict__ elog,
                                int* __restrict__ mmax, int E, int heads) {
  int t = blockIdx.x * blockDim.x + threadIdx.x;
  if (t >= E * heads) return;
  int e = t / heads;
  int h = t - e * heads;
  int s = srcE[e], d = dstE[e];
  float l = a_s[s * heads + h] + a_d[d * heads + h];
  l = (l > 0.0f) ? l : NEG_SLOPE * l;
  elog[(size_t)e * heads + h] = l;
  atomicMax(&mmax[d * heads + h], f32_to_ordkey(l));
}

// ---------------------------------------------------------------------------
// one wave per edge: ee = exp(logit - m[dst]); ssum[dst,h] += ee;
// acc[dst,:] += ee * H[src,:]   (unnormalized message accumulation)
// ---------------------------------------------------------------------------
__global__ __launch_bounds__(256) void edge_agg_kernel(
    const int* __restrict__ srcE, const int* __restrict__ dstE,
    const float* __restrict__ elog, const int* __restrict__ mmax,
    const float* __restrict__ H, float* __restrict__ ssum,
    float* __restrict__ acc, int E, int heads) {
  int wid  = (blockIdx.x * blockDim.x + threadIdx.x) >> 5;
  int lane = threadIdx.x & 31;
  if (wid >= E) return;
  int s = srcE[wid], d = dstE[wid];
  int lph  = 32 / heads;
  int head = lane / lph;
  float m  = ordkey_to_f32(mmax[d * heads + head]);
  float ee = __expf(elog[(size_t)wid * heads + head] - m);
  if ((lane % lph) == 0) atomicAdd(&ssum[d * heads + head], ee);
  float4 hv = ((const float4*)(H + (size_t)s * 128))[lane];
  float* ap = acc + (size_t)d * 128 + lane * 4;
  atomicAdd(ap + 0, ee * hv.x);
  atomicAdd(ap + 1, ee * hv.y);
  atomicAdd(ap + 2, ee * hv.z);
  atomicAdd(ap + 3, ee * hv.w);
}

// ---------------------------------------------------------------------------
// per node: out = acc/(s+1e-16) + bias -> ELU -> (+x residual) -> LayerNorm
// writes x in place. One wave per node.
// ---------------------------------------------------------------------------
__global__ __launch_bounds__(256) void node_final_kernel(
    const float* __restrict__ acc, const float* __restrict__ ssum,
    const float* __restrict__ bias, const float* __restrict__ ln_g,
    const float* __restrict__ ln_b, float* __restrict__ X,
    int N, int heads, int residual) {
  int wid  = (blockIdx.x * blockDim.x + threadIdx.x) >> 5;
  int lane = threadIdx.x & 31;
  if (wid >= N) return;
  int lph  = 32 / heads;
  int head = lane / lph;
  float denom = ssum[wid * heads + head] + 1e-16f;
  float4 a  = ((const float4*)(acc + (size_t)wid * 128))[lane];
  float4 bb = ((const float4*)bias)[lane];
  float v[4] = { a.x / denom + bb.x, a.y / denom + bb.y,
                 a.z / denom + bb.z, a.w / denom + bb.w };
#pragma unroll
  for (int j = 0; j < 4; ++j) v[j] = (v[j] > 0.0f) ? v[j] : (__expf(v[j]) - 1.0f);
  if (residual) {
    float4 xr = ((const float4*)(X + (size_t)wid * 128))[lane];
    v[0] += xr.x; v[1] += xr.y; v[2] += xr.z; v[3] += xr.w;
  }
  float s1 = v[0] + v[1] + v[2] + v[3];
  for (int msk = 1; msk < 32; msk <<= 1) s1 += __shfl_xor(s1, msk, 32);
  float mu = s1 * (1.0f / 128.0f);
  float s2 = 0.0f;
#pragma unroll
  for (int j = 0; j < 4; ++j) { float dlt = v[j] - mu; s2 += dlt * dlt; }
  for (int msk = 1; msk < 32; msk <<= 1) s2 += __shfl_xor(s2, msk, 32);
  float rs = rsqrtf(s2 * (1.0f / 128.0f) + LN_EPS);
  float4 g  = ((const float4*)ln_g)[lane];
  float4 b2 = ((const float4*)ln_b)[lane];
  float4 o;
  o.x = (v[0] - mu) * rs * g.x + b2.x;
  o.y = (v[1] - mu) * rs * g.y + b2.y;
  o.z = (v[2] - mu) * rs * g.z + b2.z;
  o.w = (v[3] - mu) * rs * g.w + b2.w;
  ((float4*)(X + (size_t)wid * 128))[lane] = o;
}

// ---------------------------------------------------------------------------
// Fused prediction MLP: [B,256] -> relu(@mw1) -> relu(@mw2) -> @mw3 -> sigmoid
// 16 batch rows per block, WMMA GEMMs through LDS; weights staged transposed.
// ---------------------------------------------------------------------------
__global__ __launch_bounds__(256) void mlp_kernel(
    const float* __restrict__ X, const int* __restrict__ uidx,
    const int* __restrict__ iidx,
    const float* __restrict__ mw1, const float* __restrict__ mb1,
    const float* __restrict__ mw2, const float* __restrict__ mb2,
    const float* __restrict__ mw3, const float* __restrict__ mb3,
    float* __restrict__ out, int B, int num_users) {
  __shared__ _Float16 A[16][264];       // combined [user||item] tile (528B stride)
  __shared__ _Float16 WT[128][136];     // transposed weights: [col][k]
  __shared__ _Float16 H1[16][136];      // hidden1 (f16)
  __shared__ float    H2[16][64];       // hidden2 (f32)

  int tid = threadIdx.x;
  int rowBase = blockIdx.x * 16;

#pragma unroll
  for (int i = 0; i < 16; ++i) {        // stage combined A: 16 x 256
    int idx = tid + i * 256;
    int r = idx >> 8, c = idx & 255;
    int gr = rowBase + r; if (gr >= B) gr = B - 1;
    int node = (c < 128) ? uidx[gr] : (iidx[gr] + num_users);
    A[r][c] = (_Float16)X[(size_t)node * 128 + (c & 127)];
  }

  int wave = tid >> 5, lane = tid & 31;
  int m = lane & 15, kb = (lane >> 4) * 8, n = lane & 15;

  v8f c1 = {};
  for (int kc = 0; kc < 4; ++kc) {      // K = 256 in 4 chunks of 64
    __syncthreads();
#pragma unroll
    for (int i = 0; i < 32; ++i) {      // stage mw1 chunk transposed: WT[col][k]
      int idx = tid + i * 256;
      int k = idx >> 7, col = idx & 127;          // k in [0,64)
      WT[col][k] = (_Float16)mw1[(size_t)(kc * 64 + k) * 128 + col];
    }
    __syncthreads();
#pragma unroll
    for (int ks = 0; ks < 2; ++ks) {
      int ka  = kc * 64 + ks * 32;      // A column base (global K)
      int kwl = ks * 32;                // WT k base (local K)
      v16h a, b;
#pragma unroll
      for (int i = 0; i < 8; ++i) {
        a[i]     = A[m][ka + kb + i];
        a[i + 8] = A[m][ka + kb + 16 + i];
        b[i]     = WT[wave * 16 + n][kwl + kb + i];
        b[i + 8] = WT[wave * 16 + n][kwl + kb + 16 + i];
      }
      c1 = __builtin_amdgcn_wmma_f32_16x16x32_f16(false, a, false, b,
                                                  (short)0, c1, false, false);
    }
  }
  __syncthreads();                      // all waves done reading A/WT

#pragma unroll
  for (int i = 0; i < 8; ++i) {         // bias + relu -> H1 (f16)
    int row = i + kb, col = wave * 16 + n;
    float v = c1[i] + mb1[col];
    H1[row][col] = (_Float16)(v > 0.0f ? v : 0.0f);
  }
#pragma unroll
  for (int i = 0; i < 32; ++i) {        // stage mw2 transposed: WT[col<64][k<128]
    int idx = tid + i * 256;            // 128*64 = 8192
    int k = idx >> 6, col = idx & 63;
    WT[col][k] = (_Float16)mw2[idx];    // mw2[k*64+col] == mw2[idx]
  }
  __syncthreads();

  if (wave < 4) {                       // 64 output cols -> waves 0..3
    v8f c2 = {};
#pragma unroll
    for (int ks = 0; ks < 4; ++ks) {    // K = 128
      int k0 = ks * 32;
      v16h a, b;
#pragma unroll
      for (int i = 0; i < 8; ++i) {
        a[i]     = H1[m][k0 + kb + i];
        a[i + 8] = H1[m][k0 + kb + 16 + i];
        b[i]     = WT[wave * 16 + n][k0 + kb + i];
        b[i + 8] = WT[wave * 16 + n][k0 + kb + 16 + i];
      }
      c2 = __builtin_amdgcn_wmma_f32_16x16x32_f16(false, a, false, b,
                                                  (short)0, c2, false, false);
    }
#pragma unroll
    for (int i = 0; i < 8; ++i) {
      int row = i + kb, col = wave * 16 + n;
      float v = c2[i] + mb2[col];
      H2[row][col] = v > 0.0f ? v : 0.0f;
    }
  }
  __syncthreads();

  int r  = tid >> 4;                    // 0..15: batch row in tile
  int cq = (tid & 15) * 4;
  float sum = H2[r][cq]     * mw3[cq]     + H2[r][cq + 1] * mw3[cq + 1] +
              H2[r][cq + 2] * mw3[cq + 2] + H2[r][cq + 3] * mw3[cq + 3];
#pragma unroll
  for (int msk = 1; msk < 16; msk <<= 1) sum += __shfl_xor(sum, msk, 32);
  if ((tid & 15) == 0) {
    int gr = rowBase + r;
    if (gr < B) {
      float z = sum + mb3[0];
      out[gr] = 1.0f / (1.0f + __expf(-z));
    }
  }
}

// ---------------------------------------------------------------------------
extern "C" void kernel_launch(void* const* d_in, const int* in_sizes, int n_in,
                              void* d_out, int out_size, void* d_ws, size_t ws_size,
                              hipStream_t stream) {
  const int*   user_idx = (const int*)d_in[0];
  const int*   item_idx = (const int*)d_in[1];
  const int*   edges    = (const int*)d_in[2];
  const float* ut       = (const float*)d_in[3];
  const float* it       = (const float*)d_in[4];
  const float* Wl[3] = { (const float*)d_in[5],  (const float*)d_in[9],  (const float*)d_in[13] };
  const float* ASl[3]= { (const float*)d_in[6],  (const float*)d_in[10], (const float*)d_in[14] };
  const float* ADl[3]= { (const float*)d_in[7],  (const float*)d_in[11], (const float*)d_in[15] };
  const float* BBl[3]= { (const float*)d_in[8],  (const float*)d_in[12], (const float*)d_in[16] };
  const float* ln_g = (const float*)d_in[17];
  const float* ln_b = (const float*)d_in[18];
  const float* mw1 = (const float*)d_in[19];
  const float* mb1 = (const float*)d_in[20];
  const float* mw2 = (const float*)d_in[21];
  const float* mb2 = (const float*)d_in[22];
  const float* mw3 = (const float*)d_in[23];
  const float* mb3 = (const float*)d_in[24];

  const int B = in_sizes[0];
  const int E = in_sizes[2] / 2;
  const int num_users = in_sizes[3] / 128;
  const int num_items = in_sizes[4] / 128;
  const int N = num_users + num_items;
  const int* srcE = edges;              // edge_index[0]
  const int* dstE = edges + E;          // edge_index[1]

  // workspace carve-out (256B aligned)
  char* wsb = (char*)d_ws;
  size_t off = 0;
  auto carve = [&](size_t bytes) -> void* {
    void* p = wsb + off;
    off += (bytes + 255) & ~(size_t)255;
    return p;
  };
  float* x    = (float*)carve((size_t)N * 128 * sizeof(float));
  float* h    = (float*)carve((size_t)N * 128 * sizeof(float));
  float* accb = (float*)carve((size_t)N * 128 * sizeof(float));
  float* a_s  = (float*)carve((size_t)N * 4 * sizeof(float));
  float* a_d  = (float*)carve((size_t)N * 4 * sizeof(float));
  float* ssum = (float*)carve((size_t)N * 4 * sizeof(float));
  int*   mmax = (int*)  carve((size_t)N * 4 * sizeof(int));
  float* elog = (float*)carve((size_t)E * 4 * sizeof(float));
  (void)ws_size; (void)n_in; (void)out_size;

  const int heads_l[3] = { 4, 4, 1 };

  {
    int total = N * 32;
    concat_kernel<<<(total + 255) / 256, 256, 0, stream>>>(ut, it, x, num_users, N);
  }

  for (int l = 0; l < 3; ++l) {
    int heads = heads_l[l];
    init_kernel<<<(N * 128 + 255) / 256, 256, 0, stream>>>(accb, ssum, mmax, N, heads);
    gemm128_kernel<<<(N + 15) / 16, 256, 0, stream>>>(x, Wl[l], h, N);
    att_coef_kernel<<<(N * 32 + 255) / 256, 256, 0, stream>>>(h, ASl[l], ADl[l],
                                                              a_s, a_d, N, heads);
    edge_max_kernel<<<(E * heads + 255) / 256, 256, 0, stream>>>(srcE, dstE, a_s, a_d,
                                                                 elog, mmax, E, heads);
    edge_agg_kernel<<<(E + 7) / 8, 256, 0, stream>>>(srcE, dstE, elog, mmax, h,
                                                     ssum, accb, E, heads);
    node_final_kernel<<<(N + 7) / 8, 256, 0, stream>>>(accb, ssum, BBl[l], ln_g, ln_b,
                                                       x, N, heads, l > 0 ? 1 : 0);
  }

  mlp_kernel<<<(B + 15) / 16, 256, 0, stream>>>(x, user_idx, item_idx,
                                                mw1, mb1, mw2, mb2, mw3, mb3,
                                                (float*)d_out, B, num_users);
}
